// BaseAttention_66975720013872
// MI455X (gfx1250) — matrix operand order repaired and checked by reading, not compiled, not run on hardware
//
#include <hip/hip_runtime.h>

#define NB 2
#define NH 16
#define T_LEN 2048
#define DH 64
#define WINDOW 512
// SCALE * log2(e): scores kept in base-2 units throughout the softmax
#define C2SCALE 0.1803368801111f
#define NEGBIG -1e30f

#define BQ 64                   // q rows per workgroup
#define TQ 16                   // q rows per wave
#define TK 32                   // keys per LDS tile
#define NWAVES 4
#define NTHREADS 128

#define KS_LD 72                // ushort row stride for Ks/Vs (64 dims + pad, 16B aligned)
#define QT_LD 24                // ushort row stride for Qt (16 q cols + pad, 16B aligned)

typedef __attribute__((ext_vector_type(16))) __bf16 v16bf;
typedef __attribute__((ext_vector_type(8)))  float  v8f;

union V16 {
  v16bf v;
  uint4 q[2];
  unsigned int u[8];
  unsigned short h[16];
};

// fp32 -> bf16 with round-half-up (1 add + shift); finite inputs only
__device__ __forceinline__ unsigned short f2bf(float f) {
  return (unsigned short)((__float_as_uint(f) + 0x8000u) >> 16);
}

// two fp32 -> packed bf16x2: bias-add then one v_perm_b32 picking the two
// high halves: D = { b[31:16], a[31:16] }
__device__ __forceinline__ unsigned int cvt2(float a, float b) {
  unsigned int ua = __float_as_uint(a) + 0x8000u;
  unsigned int ub = __float_as_uint(b) + 0x8000u;
  return __builtin_amdgcn_perm(ub, ua, 0x07060302u);
}

__device__ __forceinline__ uint4 pack8(float4 a, float4 b) {
  uint4 o;
  o.x = cvt2(a.x, a.y);
  o.y = cvt2(a.z, a.w);
  o.z = cvt2(b.x, b.y);
  o.w = cvt2(b.z, b.w);
  return o;
}

__device__ __forceinline__ v8f wmma_bf16(v16bf a, v16bf b, v8f c) {
  // D = A(16x32 bf16) * B(32x16 bf16) + C(16x16 f32)
  return __builtin_amdgcn_wmma_f32_16x16x32_bf16(
      /*neg_a=*/false, a, /*neg_b=*/false, b,
      /*c_mod=*/(short)0, c, /*reuse_a=*/false, /*reuse_b=*/false);
}

__global__ __launch_bounds__(NTHREADS)
void swa_flash_kernel(const float* __restrict__ Qg,
                      const float* __restrict__ Kg,
                      const float* __restrict__ Vg,
                      float* __restrict__ Og) {
  __shared__ __align__(16) unsigned short Ks[TK * KS_LD];
  __shared__ __align__(16) unsigned short Vs[TK * KS_LD];
  __shared__ __align__(16) unsigned short Qt[NWAVES][DH * QT_LD];

  const int tid  = threadIdx.x;
  const int wid  = tid >> 5;
  const int lane = tid & 31;
  const int lrow = lane & 15;           // q column index in S^T D-layout
  const int hi   = lane >> 4;           // lane half
  const int c0   = hi << 3;             // 0 or 8: A-operand K-chunk base / C-row base

  const int qb = blockIdx.x;
  const int h  = blockIdx.y;
  const int b  = blockIdx.z;
  const int bh = b * NH + h;
  const int q0 = qb * BQ;

  const size_t baseBH = (size_t)bh * T_LEN * DH;

  // ---------- stage this wave's 16 Q rows, transposed, fp32->bf16 ----------
  {
    const int r     = lrow;
    const int dbase = hi * 32;          // lanes 0-15: dims 0-31, lanes 16-31: 32-63
    const float4* qp =
        (const float4*)(Qg + baseBH + (size_t)(q0 + wid * TQ + r) * DH + dbase);
#pragma unroll
    for (int i = 0; i < 8; ++i) {
      float4 f = qp[i];
      int d = dbase + i * 4;
      Qt[wid][(d + 0) * QT_LD + r] = f2bf(f.x);
      Qt[wid][(d + 1) * QT_LD + r] = f2bf(f.y);
      Qt[wid][(d + 2) * QT_LD + r] = f2bf(f.z);
      Qt[wid][(d + 3) * QT_LD + r] = f2bf(f.w);
    }
  }

  // B-operands of S^T = K*Q^T (loop invariant): lane = q column, elements = dims in table order
  V16 bq0, bq1;
  {
#pragma unroll
    for (int s = 0; s < 16; ++s) {
      const int kk = (s < 8) ? (c0 + s) : (16 + c0 + s - 8);
      bq0.h[s] = Qt[wid][kk * QT_LD + lrow];
      bq1.h[s] = Qt[wid][(32 + kk) * QT_LD + lrow];
    }
  }

  // ---------- key tile range (uniform across block) ----------
  int kt_lo = q0 - (WINDOW - 1);
  if (kt_lo < 0) kt_lo = 0;
  kt_lo &= ~(TK - 1);
  const int ntiles = ((q0 + BQ - 1 - kt_lo) >> 5) + 1;
  // tiles with kt in [kt_midlo, q0) are fully valid for every row in the block
  int kt_midlo = q0 - (WINDOW - BQ);    // q0 - 448
  if (kt_midlo < kt_lo) kt_midlo = kt_lo;

  // staging coordinates: 128 threads cover a 32x64 tile, 16 floats each
  const int sr = tid >> 2;              // key row 0..31
  const int sc = (tid & 3) * 16;        // dim segment

  float4 kr[4], vr[4];
  {
    const float4* kp = (const float4*)(Kg + baseBH + (size_t)(kt_lo + sr) * DH + sc);
    const float4* vp = (const float4*)(Vg + baseBH + (size_t)(kt_lo + sr) * DH + sc);
#pragma unroll
    for (int i = 0; i < 4; ++i) { kr[i] = kp[i]; vr[i] = vp[i]; }
  }

  // ---------- flash state ----------
  const v8f vzero = {};
  float mrun = NEGBIG;                  // base-2 units
  float lrun = 0.0f;
  v8f oacc[4];
#pragma unroll
  for (int t = 0; t < 4; ++t) oacc[t] = vzero;

  const int qrow = q0 + wid * TQ + lrow;

  int kt = kt_lo;
  for (int it = 0; it < ntiles; ++it) {
    __syncthreads();                    // previous tile's consumers are done
    {
      unsigned short* kd = &Ks[sr * KS_LD + sc];
      unsigned short* vd = &Vs[sr * KS_LD + sc];
      *(uint4*)(kd)     = pack8(kr[0], kr[1]);
      *(uint4*)(kd + 8) = pack8(kr[2], kr[3]);
      *(uint4*)(vd)     = pack8(vr[0], vr[1]);
      *(uint4*)(vd + 8) = pack8(vr[2], vr[3]);
    }
    __syncthreads();

    // prefetch next K/V tile into registers; loads overlap this tile's WMMAs
    if (it + 1 < ntiles) {
      const int ktn = kt + TK;
      const float4* kp = (const float4*)(Kg + baseBH + (size_t)(ktn + sr) * DH + sc);
      const float4* vp = (const float4*)(Vg + baseBH + (size_t)(ktn + sr) * DH + sc);
#pragma unroll
      for (int i = 0; i < 4; ++i) { kr[i] = kp[i]; vr[i] = vp[i]; }
    }

    // ---- S^T = K * Q^T : 2 key groups x (2 WMMA over D=64) ----
    float st[2][8];
#pragma unroll
    for (int g = 0; g < 2; ++g) {
      const unsigned short* kp = &Ks[(g * 16 + lrow) * KS_LD];
      V16 a0, a1;                       // A-operand: lane holds K row, dim chunks
      a0.q[0] = *(const uint4*)(kp + c0);
      a0.q[1] = *(const uint4*)(kp + 16 + c0);
      a1.q[0] = *(const uint4*)(kp + 32 + c0);
      a1.q[1] = *(const uint4*)(kp + 48 + c0);
      v8f acc = vzero;
      acc = wmma_bf16(a0.v, bq0.v, acc);
      acc = wmma_bf16(a1.v, bq1.v, acc);
#pragma unroll
      for (int i = 0; i < 8; ++i) st[g][i] = acc[i];
    }

    // ---- masked online softmax (base-2); lane holds row `qrow` ----
    // block-uniform: only window-lower-edge and diagonal tiles need masking
    const bool masked = (kt < kt_midlo) || (kt + TK > q0);
    float sv[16];
    if (masked) {
#pragma unroll
      for (int g = 0; g < 2; ++g)
#pragma unroll
        for (int i = 0; i < 8; ++i) {
          int key = kt + g * 16 + c0 + i;
          float x = st[g][i] * C2SCALE;
          // key<=q && q-key<W  <=>  (unsigned)(q-key) < W
          sv[g * 8 + i] = ((unsigned)(qrow - key) < (unsigned)WINDOW) ? x : NEGBIG;
        }
    } else {
#pragma unroll
      for (int g = 0; g < 2; ++g)
#pragma unroll
        for (int i = 0; i < 8; ++i)
          sv[g * 8 + i] = st[g][i] * C2SCALE;
    }

    float mt = NEGBIG;
#pragma unroll
    for (int j = 0; j < 16; ++j) mt = fmaxf(mt, sv[j]);
    mt = fmaxf(mt, __shfl_xor(mt, 16, 32));
    const float mnew  = fmaxf(mrun, mt);
    const float alpha = exp2f(mrun - mnew);

    float p[16];
    float lt = 0.0f;
#pragma unroll
    for (int j = 0; j < 16; ++j) {
      p[j] = exp2f(sv[j] - mnew);
      lt += p[j];
    }
    V16 ap;                             // P already in A-operand layout
#pragma unroll
    for (int j = 0; j < 8; ++j) ap.u[j] = cvt2(p[2 * j], p[2 * j + 1]);
    lt += __shfl_xor(lt, 16, 32);
    lrun = lrun * alpha + lt;
    mrun = mnew;

    // ---- rescale O (alpha broadcast to O's lane layout) and do P*V ----
#pragma unroll
    for (int vv = 0; vv < 8; ++vv) {
      float ab = __shfl(alpha, vv + c0, 32);
#pragma unroll
      for (int t = 0; t < 4; ++t) oacc[t][vv] *= ab;
    }
#pragma unroll
    for (int t = 0; t < 4; ++t) {
      const unsigned short* vp = &Vs[t * 16 + lrow];
      V16 bv;                           // B-operand: lane = dim, elements = keys in table order
#pragma unroll
      for (int s = 0; s < 16; ++s) {
        const int kk = (s < 8) ? (c0 + s) : (16 + c0 + s - 8);
        bv.h[s] = vp[kk * KS_LD];
      }
      oacc[t] = wmma_bf16(ap.v, bv.v, oacc[t]);
    }

    kt += TK;
  }

  // ---------- normalize and store ----------
  float* op = Og + baseBH;
#pragma unroll
  for (int vv = 0; vv < 8; ++vv) {
    float lb  = __shfl(lrun, vv + c0, 32);
    float inv = 1.0f / lb;
    const int row = q0 + wid * TQ + c0 + vv;
#pragma unroll
    for (int t = 0; t < 4; ++t) {
      op[(size_t)row * DH + t * 16 + lrow] = oacc[t][vv] * inv;
    }
  }
}

extern "C" void kernel_launch(void* const* d_in, const int* in_sizes, int n_in,
                              void* d_out, int out_size, void* d_ws, size_t ws_size,
                              hipStream_t stream) {
  (void)in_sizes; (void)n_in; (void)out_size; (void)d_ws; (void)ws_size;
  const float* q = (const float*)d_in[0];
  const float* k = (const float*)d_in[1];
  const float* v = (const float*)d_in[2];
  float* o = (float*)d_out;
  dim3 grid(T_LEN / BQ, NH, NB);
  dim3 block(NTHREADS);
  hipLaunchKernelGGL(swa_flash_kernel, grid, block, 0, stream, q, k, v, o);
}
